// BahdanauAttention_13675175870746
// MI455X (gfx1250) — matrix operand order repaired and checked by reading, not compiled, not run on hardware
//
#include <hip/hip_runtime.h>
#include <stdint.h>
#include <math.h>

// ---------------------------------------------------------------------------
// Bahdanau attention, MI455X (gfx1250, wave32).
//   B=32, S=2048, H=1024, F=2H=2048.
// Pipeline:
//   1) pack_ua:   Ua_w [2048,1024] f32 -> bf16 B-fragments (WMMA 32x16 layout)
//   2) qproj:     q[b][h] = dec@Wa + Wa_b + Ua_b   (biases folded)
//   3) scores:    fused  score[b,s] = sum_h Va[h]*tanh(q[b,h] + (enc@Ua)[b,s,h])
//                 via v_wmma_f32_16x16x32_bf16, 64x512 tile per workgroup,
//                 double-buffered LDS staging, NT hints on the enc stream
//   4) softmax over S
//   5) context:   attn^T @ enc (memory-bound streaming pass, NT loads)
// ---------------------------------------------------------------------------

typedef __attribute__((ext_vector_type(16))) __bf16 v16bf;
typedef __attribute__((ext_vector_type(8)))  __bf16 v8bf;
typedef __attribute__((ext_vector_type(8)))  float  v8f;
typedef __attribute__((ext_vector_type(4)))  float  fvec4;

#define Bn 32
#define Sn 2048
#define Hn 1024
#define Fn 2048   // 2H

static __device__ __forceinline__ uint32_t pack_bf16(float lo, float hi) {
  // round-to-nearest-even f32 -> bf16, packed pair (low = even K)
  uint32_t a = __float_as_uint(lo);
  uint32_t b = __float_as_uint(hi);
  a = (a + 0x7FFFu + ((a >> 16) & 1u)) >> 16;
  b = (b + 0x7FFFu + ((b >> 16) & 1u)) >> 16;
  return (a & 0xFFFFu) | (b << 16);
}

// -------- 1) Pack Ua into WMMA B-fragment order: [ktile][ntile][lane][8 dw] --
// B 32x16 bf16 packing used here (contiguous k, matches A): lane n = L%16, khalf = L/16, dword d holds
// K = khalf*16 + 2d (low) and +1 (high).
__global__ __launch_bounds__(256) void pack_ua_kernel(
    const float* __restrict__ Ua, uint32_t* __restrict__ bp) {
  uint32_t t = blockIdx.x * 256u + threadIdx.x;      // 0 .. 64*64*32*8-1
  uint32_t d     = t & 7u;
  uint32_t lane  = (t >> 3) & 31u;
  uint32_t ntile = (t >> 8) & 63u;
  uint32_t ktile = t >> 14;
  uint32_t khalf = lane >> 4;
  uint32_t n  = ntile * 16u + (lane & 15u);
  uint32_t k0 = ktile * 32u + khalf * 16u + 2u * d;
  bp[t] = pack_bf16(Ua[(size_t)k0 * Hn + n], Ua[(size_t)(k0 + 1) * Hn + n]);
}

// -------- 2) q[b][h] = dec[b]@Wa[:,h] + Wa_b[h] + Ua_b[h] -------------------
__global__ __launch_bounds__(256) void qproj_kernel(
    const float* __restrict__ dec, const float* __restrict__ Wa,
    const float* __restrict__ Wab, const float* __restrict__ Uab,
    float* __restrict__ q) {
  int id = blockIdx.x * 256 + threadIdx.x;   // 0..32767
  int b = id >> 10, h = id & 1023;
  const float* dv = dec + (size_t)b * Hn;
  float acc = Wab[h] + Uab[h];
  for (int k = 0; k < Hn; ++k) acc += dv[k] * Wa[(size_t)k * Hn + h];
  q[id] = acc;
}

// -------- 3) fused scores GEMM: 64 rows x 512 cols per WG, K=2048 -----------
__global__ __launch_bounds__(256) void bahdanau_scores_kernel(
    const float* __restrict__ enc,       // [B*S, F] f32
    const uint32_t* __restrict__ bpack,  // [64][64][32][8] bf16 pairs
    const float* __restrict__ qb,        // [B, H] (biases folded in)
    const float* __restrict__ va,        // [H]
    float* __restrict__ scores)          // [B*S], pre-zeroed
{
  // double-buffered A staging: 64 rows x 16 dwords (+8 pad for banks), x2
  __shared__ __align__(16) uint32_t ldsA[2][64 * 24];
  __shared__ float ldsScore[64];

  const int tid     = threadIdx.x;
  const int lane    = tid & 31;
  const int wv      = tid >> 5;               // wave 0..7
  const int rowBase = blockIdx.x * 64;        // flattened (b,s) row
  const int colHalf = blockIdx.y;             // 0..1 -> cols [colHalf*512, +512)
  const int b       = rowBase >> 11;          // 64 | 2048 => single b per WG

  if (tid < 64) ldsScore[tid] = 0.0f;

  const int laneN = lane & 15;
  const int khalf = lane >> 4;

  v8f acc[4][4];
  #pragma unroll
  for (int mt = 0; mt < 4; ++mt)
    #pragma unroll
    for (int nt = 0; nt < 4; ++nt) acc[mt][nt] = (v8f){};

  const float* encBase = enc + (size_t)rowBase * Fn;
  const int fillRow = tid >> 2;   // 0..63
  const int fillG   = tid & 3;    // 0..3 (8 floats each)
  const fvec4* fillSrc =
      (const fvec4*)(encBase + (size_t)fillRow * Fn + fillG * 8);

  // prologue: load chunk 0 (non-temporal: enc is a single-use 512MB stream)
  fvec4 f0 = __builtin_nontemporal_load(fillSrc + 0);
  fvec4 f1 = __builtin_nontemporal_load(fillSrc + 1);

  for (int kc = 0; kc < 64; ++kc) {           // K chunks of 32
    const int buf = kc & 1;
    // store current chunk into its LDS buffer
    uint32_t* dst = &ldsA[buf][fillRow * 24 + fillG * 4];
    dst[0] = pack_bf16(f0.x, f0.y);
    dst[1] = pack_bf16(f0.z, f0.w);
    dst[2] = pack_bf16(f1.x, f1.y);
    dst[3] = pack_bf16(f1.z, f1.w);
    // issue next chunk's global load before the barrier (overlaps with WMMA)
    if (kc < 63) {
      const fvec4* nsrc = (const fvec4*)((const float*)fillSrc + (kc + 1) * 32);
      f0 = __builtin_nontemporal_load(nsrc + 0);
      f1 = __builtin_nontemporal_load(nsrc + 1);
    }
    __syncthreads();   // buffer `buf` filled; previous buffer free for kc+1

    // --- A fragments (16x32 bf16): v0-7 <- K khalf*16+0..15 ---
    v16bf afrag[4];
    #pragma unroll
    for (int mt = 0; mt < 4; ++mt) {
      int m = mt * 16 + laneN;
      v8bf lo = *(const v8bf*)&ldsA[buf][m * 24 + khalf * 8];
      v8bf hi = *(const v8bf*)&ldsA[buf][m * 24 + 4 + khalf * 8];
      afrag[mt] = __builtin_shufflevector(lo, hi,
          0,1,2,3,4,5,6,7,8,9,10,11,12,13,14,15);
    }

    // --- B fragments (pre-packed, 32B/lane contiguous) + 16 WMMAs ---
    #pragma unroll
    for (int nt = 0; nt < 4; ++nt) {
      int ntile = colHalf * 32 + wv * 4 + nt;
      const v16bf* bp = (const v16bf*)(bpack +
          ((size_t)(kc * 64 + ntile) * 32 + lane) * 8);
      v16bf bfrag = *bp;
      #pragma unroll
      for (int mt = 0; mt < 4; ++mt)
        acc[mt][nt] = __builtin_amdgcn_wmma_f32_16x16x32_bf16(
            false, afrag[mt], false, bfrag, (short)0, acc[mt][nt],
            false, false);
    }
  }

  // --- fused epilogue: score += Va[col] * tanh(k + q) ---
  // C layout: col = laneN (fixed per lane), row-in-tile = khalf*8 + v
  float psum[4][8];
  #pragma unroll
  for (int mt = 0; mt < 4; ++mt)
    #pragma unroll
    for (int v = 0; v < 8; ++v) psum[mt][v] = 0.0f;

  #pragma unroll
  for (int nt = 0; nt < 4; ++nt) {
    int col   = colHalf * 512 + (wv * 4 + nt) * 16 + laneN;
    float qv  = qb[b * Hn + col];
    float vav = va[col];
    #pragma unroll
    for (int mt = 0; mt < 4; ++mt)
      #pragma unroll
      for (int v = 0; v < 8; ++v)
        psum[mt][v] += vav * tanhf(acc[mt][nt][v] + qv);
  }
  #pragma unroll
  for (int mt = 0; mt < 4; ++mt)
    #pragma unroll
    for (int v = 0; v < 8; ++v)
      atomicAdd(&ldsScore[mt * 16 + khalf * 8 + v], psum[mt][v]);
  __syncthreads();
  if (tid < 64) atomicAdd(&scores[rowBase + tid], ldsScore[tid]);
}

// -------- 4) softmax over S per batch row -----------------------------------
__global__ __launch_bounds__(256) void softmax_kernel(
    const float* __restrict__ scores, float* __restrict__ attn) {
  __shared__ float red[256];
  int b = blockIdx.x, t = threadIdx.x;
  const float* row = scores + (size_t)b * Sn;
  float m = -1e30f;
  for (int s = t; s < Sn; s += 256) m = fmaxf(m, row[s]);
  red[t] = m; __syncthreads();
  for (int o = 128; o > 0; o >>= 1) {
    if (t < o) red[t] = fmaxf(red[t], red[t + o]);
    __syncthreads();
  }
  m = red[0]; __syncthreads();
  float sum = 0.0f;
  for (int s = t; s < Sn; s += 256) sum += __expf(row[s] - m);
  red[t] = sum; __syncthreads();
  for (int o = 128; o > 0; o >>= 1) {
    if (t < o) red[t] += red[t + o];
    __syncthreads();
  }
  float inv = 1.0f / red[0];
  for (int s = t; s < Sn; s += 256)
    attn[(size_t)b * Sn + s] = __expf(row[s] - m) * inv;
}

// -------- 5) context[b][f] = sum_s attn[b,s] * enc[b,s,f] -------------------
__global__ __launch_bounds__(256) void context_kernel(
    const float* __restrict__ enc, const float* __restrict__ attn,
    float* __restrict__ ctx) {
  int t = threadIdx.x;
  int f = blockIdx.x * 256 + t;       // 8 tiles of 256 cols
  int b = blockIdx.y;
  int s0 = blockIdx.z * 256;          // 8 chunks over S
  const float* e = enc + ((size_t)b * Sn + s0) * Fn + f;
  const float* w = attn + (size_t)b * Sn + s0;
  float acc = 0.0f;
  for (int s = 0; s < 256; ++s)
    acc += w[s] * __builtin_nontemporal_load(e + (size_t)s * Fn);
  atomicAdd(&ctx[(size_t)b * Fn + f], acc);
}

// ---------------------------------------------------------------------------
extern "C" void kernel_launch(void* const* d_in, const int* in_sizes, int n_in,
                              void* d_out, int out_size, void* d_ws, size_t ws_size,
                              hipStream_t stream) {
  (void)in_sizes; (void)n_in; (void)out_size; (void)ws_size;
  const float* dec  = (const float*)d_in[0];   // [B,H]
  const float* enc  = (const float*)d_in[1];   // [B,S,2H]
  const float* Wa_w = (const float*)d_in[2];   // [H,H]
  const float* Wa_b = (const float*)d_in[3];   // [H]
  const float* Ua_w = (const float*)d_in[4];   // [2H,H]
  const float* Ua_b = (const float*)d_in[5];   // [H]
  const float* Va_w = (const float*)d_in[6];   // [H,1]
  // d_in[7] = Va_b: constant shift over S, cancels in softmax.

  float* out  = (float*)d_out;
  float* ctx  = out;                 // [B, 2H] = 65536 floats
  float* attn = out + 65536;         // [B, S]  = 65536 floats

  float*    q      = (float*)d_ws;                 // 32768 floats
  float*    scores = q + 32768;                    // 65536 floats
  uint32_t* bpack  = (uint32_t*)(scores + 65536);  // 1,048,576 dwords (4 MB)

  hipMemsetAsync(scores, 0, 65536 * sizeof(float), stream);
  hipMemsetAsync(ctx,    0, 65536 * sizeof(float), stream);

  pack_ua_kernel<<<4096, 256, 0, stream>>>(Ua_w, bpack);
  qproj_kernel<<<128, 256, 0, stream>>>(dec, Wa_w, Wa_b, Ua_b, q);
  bahdanau_scores_kernel<<<dim3(1024, 2), 256, 0, stream>>>(
      enc, bpack, q, Va_w, scores);
  softmax_kernel<<<32, 256, 0, stream>>>(scores, attn);
  context_kernel<<<dim3(8, 32, 8), 256, 0, stream>>>(enc, attn, ctx);
}
